// ESN_57698590654506
// MI455X (gfx1250) — matrix operand order repaired and checked by reading, not compiled, not run on hardware
//
#include <hip/hip_runtime.h>
#include <hip/hip_bf16.h>

// ============================================================================
// ESN recurrence for MI455X (gfx1250, wave32, WMMA).
//
// Parallel structure: recurrence is independent per batch row -> 16 persistent
// workgroups (one per 16-row batch tile), 16 waves each; wave w owns n-tiles
// {2w, 2w+1} for the whole time loop. Per step, per wave: K=576 reduction
// (512 recurrent + 64 input) as 18 k-blocks of V_WMMA_F32_16X16X32_BF16,
// accumulating in f32. State lives in registers (D-fragment layout); shared
// across waves each step via double-buffered LDS + one barrier.
// Nonlinearity is a branchless tanh (exp + rcp) so the serial critical path
// stays WMMA/LDS-bound instead of VALU/branch-bound.
// ============================================================================

#define B_     256
#define T_     512
#define D_     64
#define N_     512
#define LEAK   0.5f
#define KTOT   576          // N_ + D_
#define KB     18           // KTOT / 32
#define NT     32           // N_ / 16
#define STRIDE 584          // 576 + 8 pad: stride/2 dwords = 292 = 4*73 -> rows
                            // land on distinct 4-dword bank groups (mod 64)

typedef __attribute__((ext_vector_type(16))) __bf16       v16bf;
typedef __attribute__((ext_vector_type(8)))  float        v8f;
typedef __attribute__((ext_vector_type(4)))  unsigned int v4u;

union Frag { v16bf v; v4u u[2]; };

// f32 -> bf16, round-to-nearest-even
static __device__ __forceinline__ unsigned short f2bf(float f) {
  unsigned u = __float_as_uint(f);
  u += 0x7fffu + ((u >> 16) & 1u);
  return (unsigned short)(u >> 16);
}

// Branchless tanh: sign(x) * (1-e)/(1+e), e = exp(-2|x|) in (0,1].
// No overflow/NaN paths -> no exec-mask divergence; ~7 ops via v_exp/v_rcp.
// Error ~2 ulp, far below the bf16 quantization already applied to the state.
static __device__ __forceinline__ float fast_tanh(float x) {
  const float ax = __builtin_fabsf(x);
  const float e  = __expf(-2.0f * ax);
  const float r  = __fdividef(1.0f - e, 1.0f + e);
  return __builtin_copysignf(r, x);
}

// ----------------------------------------------------------------------------
// Pack W_res (512x512) and W_in (64x512) row-major f32 into bf16 B-fragment
// layout: frag(tile, kb) = 32 lanes x 8 dwords. k order used here (contiguous per lane half, same as A):
//   lane l<16 : column n = tile*16 + l, vgpr j holds (K=2j, K=2j+1)   [K 0..15]
//   lane l>=16: same column,           vgpr j holds (K=16+2j, 17+2j)  [K 16..31]
// Fragments for one tile are contiguous in kb -> each wave streams 18 KB/tile.
// ----------------------------------------------------------------------------
__global__ void esn_pack(const float* __restrict__ W_in,
                         const float* __restrict__ W_res,
                         unsigned int* __restrict__ WP) {
  const int tile  = blockIdx.x;            // 0..31
  const int kb    = blockIdx.y;            // 0..17
  const int lane  = threadIdx.x;           // 0..31 (one wave)
  const int n     = tile * 16 + (lane & 15);
  const int kbase = kb * 32 + ((lane < 16) ? 0 : 16);
  unsigned int* dst = WP + ((size_t)(tile * KB + kb) * 32 + lane) * 8;
#pragma unroll
  for (int j = 0; j < 8; ++j) {
    const int k0 = kbase + 2 * j;
    const int k1 = k0 + 1;
    const float w0 = (k0 < N_) ? W_res[(size_t)k0 * N_ + n] : W_in[(size_t)(k0 - N_) * N_ + n];
    const float w1 = (k1 < N_) ? W_res[(size_t)k1 * N_ + n] : W_in[(size_t)(k1 - N_) * N_ + n];
    dst[j] = (unsigned)f2bf(w0) | ((unsigned)f2bf(w1) << 16);
  }
}

// ----------------------------------------------------------------------------
// Persistent recurrence kernel. grid = 16 (batch tiles), block = 512 (16 waves)
// ----------------------------------------------------------------------------
__global__ __launch_bounds__(512, 1)
void esn_recur(const float* __restrict__ inputs,
               const float* __restrict__ b_in,
               const float* __restrict__ b_res,
               const unsigned int* __restrict__ WP,
               float* __restrict__ out) {
  // rows 0..15 = batch rows of this tile; cols [0,512) = x (bf16),
  // cols [512,576) = current input tile (bf16). Double buffered.
  __shared__ unsigned short xbuf[2][16][STRIDE];

  const int tid  = threadIdx.x;
  const int lane = tid & 31;
  const int wave = tid >> 5;
  const int half = lane >> 4;        // 0: lanes 0-15, 1: lanes 16-31
  const int nl   = lane & 15;
  const int b0   = blockIdx.x * 16;

  const int tA = wave * 2, tB = wave * 2 + 1;      // this wave's n-tiles
  const int nA = tA * 16 + nl, nB = tB * 16 + nl;  // this lane's columns
  const float binA = b_in[nA], brA = b_res[nA];
  const float binB = b_in[nB], brB = b_res[nB];

  // A-frag (16x32 bf16) per-lane addressing: row m = nl; within k-block,
  // vgprs 0-3 cover K = half*16 + {0..7}, vgprs 4-7 cover K = half*16 + 8 + {0..7}
  const int eoff = half * 16;

  const unsigned int* wpA = WP + (size_t)(tA * KB) * 256 + lane * 8;
  const unsigned int* wpB = WP + (size_t)(tB * KB) * 256 + lane * 8;

  float*       Xout  = out;                          // (B, T, N)
  float*       Xlast = out + (size_t)B_ * T_ * N_;   // (B, N)

  // Zero the x region of buffer 0 (makes step 0 uniform: x_{-1} = 0), and
  // load+convert the input tile for t=0.
  for (int i = tid; i < 16 * 512; i += 512) xbuf[0][i >> 9][i & 511] = 0;
  {
    const int e = tid * 2, m = e >> 6, k = e & 63;      // 16*64 = 1024 elems
    const float2 v = *(const float2*)&inputs[((size_t)(b0 + m) * T_ + 0) * D_ + k];
    xbuf[0][m][512 + k]     = f2bf(v.x);
    xbuf[0][m][512 + k + 1] = f2bf(v.y);
  }
  __syncthreads();

  v8f xA = {0.f,0.f,0.f,0.f,0.f,0.f,0.f,0.f};
  v8f xB = xA;

  for (int t = 0; t < T_; ++t) {
    const int cur = t & 1, nxt = cur ^ 1;
    // t==0: x = a*tanh(U0 + b_in) -> bias is b_in only, x-region is zero.
    const float biasA = (t == 0) ? binA : (binA + brA);
    const float biasB = (t == 0) ? binB : (binB + brB);
    v8f accA = {biasA,biasA,biasA,biasA,biasA,biasA,biasA,biasA};
    v8f accB = {biasB,biasB,biasB,biasB,biasB,biasB,biasB,biasB};

    const unsigned short* row = &xbuf[cur][nl][0];

#pragma unroll 3
    for (int kb = 0; kb < KB; ++kb) {
      Frag a, bA, bB;
      const int e0 = kb * 32 + eoff;
      a.u[0]  = *(const v4u*)(row + e0);
      a.u[1]  = *(const v4u*)(row + e0 + 8);
      bA.u[0] = *(const v4u*)(wpA + kb * 256);
      bA.u[1] = *(const v4u*)(wpA + kb * 256 + 4);
      bB.u[0] = *(const v4u*)(wpB + kb * 256);
      bB.u[1] = *(const v4u*)(wpB + kb * 256 + 4);
      accA = __builtin_amdgcn_wmma_f32_16x16x32_bf16(
                 false, a.v, false, bA.v, (short)0, accA, false, false);
      accB = __builtin_amdgcn_wmma_f32_16x16x32_bf16(
                 false, a.v, false, bB.v, (short)0, accB, false, false);
    }

    // Leaky integration in registers: x_t = (1-a)*x_{t-1} + a*tanh(acc)
#pragma unroll
    for (int r = 0; r < 8; ++r) {
      xA[r] = (1.0f - LEAK) * xA[r] + LEAK * fast_tanh(accA[r]);
      xB[r] = (1.0f - LEAK) * xB[r] + LEAK * fast_tanh(accB[r]);
    }

    // Store X (coalesced: 16 lanes -> 16 consecutive n), share state via LDS.
#pragma unroll
    for (int r = 0; r < 8; ++r) {
      const int m = r + half * 8;                        // D-layout row
      const size_t obase = ((size_t)(b0 + m) * T_ + t) * N_;
      Xout[obase + nA] = xA[r];
      Xout[obase + nB] = xB[r];
      xbuf[nxt][m][nA] = f2bf(xA[r]);
      xbuf[nxt][m][nB] = f2bf(xB[r]);
    }

    // Cooperative load+convert of next step's input tile into the nxt buffer;
    // prefetch the t+2 tile toward L2 (global_prefetch_b8).
    if (t + 1 < T_) {
      const int e = tid * 2, m = e >> 6, k = e & 63;
      const float* src = &inputs[((size_t)(b0 + m) * T_ + (t + 1)) * D_ + k];
      const float2 v = *(const float2*)src;
      xbuf[nxt][m][512 + k]     = f2bf(v.x);
      xbuf[nxt][m][512 + k + 1] = f2bf(v.y);
      if (t + 2 < T_)
        __builtin_prefetch(&inputs[((size_t)(b0 + m) * T_ + (t + 2)) * D_ + k], 0, 0);
    }
    __syncthreads();
  }

  // Final state X[:, -1, :] (still live in registers).
#pragma unroll
  for (int r = 0; r < 8; ++r) {
    const int m = r + half * 8;
    Xlast[(size_t)(b0 + m) * N_ + nA] = xA[r];
    Xlast[(size_t)(b0 + m) * N_ + nB] = xB[r];
  }
}

// ============================================================================
extern "C" void kernel_launch(void* const* d_in, const int* in_sizes, int n_in,
                              void* d_out, int out_size, void* d_ws, size_t ws_size,
                              hipStream_t stream) {
  (void)in_sizes; (void)n_in; (void)out_size; (void)ws_size;
  const float* inputs = (const float*)d_in[0];   // (B,T,D)
  const float* W_in   = (const float*)d_in[1];   // (D,N)
  const float* b_in   = (const float*)d_in[2];   // (N)
  const float* W_res  = (const float*)d_in[3];   // (N,N)
  const float* b_res  = (const float*)d_in[4];   // (N)

  unsigned int* WP = (unsigned int*)d_ws;        // 576 KB packed bf16 weights

  esn_pack<<<dim3(NT, KB), 32, 0, stream>>>(W_in, W_res, WP);
  esn_recur<<<dim3(B_ / 16), 512, 0, stream>>>(inputs, b_in, b_res, WP,
                                               (float*)d_out);
}
